// RWKV_TimeMix_6536940224793
// MI455X (gfx1250) — matrix operand order repaired and checked by reading, not compiled, not run on hardware
//
#include <hip/hip_runtime.h>

// ---------------------------------------------------------------------------
// RWKV TimeMix step on gfx1250 (MI455X).
//  - 4x GEMM 8192x2048x2048 via v_wmma_f32_16x16x32_f16 (f32 accumulate)
//  - double-buffered GLOBAL_LOAD_ASYNC_TO_LDS_B128 staging (ASYNCcnt)
//  - elementwise mix / WKV kernels for the exp/max state math
// Outputs (fp32, concat): rwkv | new_A | new_B | new_p | new_x  (each B*C)
// ---------------------------------------------------------------------------

typedef __attribute__((ext_vector_type(16))) _Float16 v16h;
typedef __attribute__((ext_vector_type(8)))  _Float16 v8h;
typedef __attribute__((ext_vector_type(4)))  _Float16 v4h;
typedef __attribute__((ext_vector_type(8)))  float    v8f;
typedef __attribute__((ext_vector_type(4)))  float    v4f;

#define B_DIM 8192
#define C_DIM 2048
static constexpr size_t BC  = (size_t)B_DIM * C_DIM;   // 16,777,216
static constexpr size_t CxC = (size_t)C_DIM * C_DIM;   //  4,194,304

// ---- token-shift mix: out_f16 = x*m + sx*(1-m) ------------------------------
__global__ __launch_bounds__(256) void mix_one_kernel(
    const float* __restrict__ x, const float* __restrict__ sx,
    const float* __restrict__ tm, _Float16* __restrict__ out)
{
  size_t i = ((size_t)blockIdx.x * blockDim.x + threadIdx.x) * 4;
  int c = (int)(i & (size_t)(C_DIM - 1));
  v4f xv = *(const v4f*)(x + i);
  v4f sv = *(const v4f*)(sx + i);
  v4f mv = *(const v4f*)(tm + c);
  v4h o;
#pragma unroll
  for (int j = 0; j < 4; ++j)
    o[j] = (_Float16)(sv[j] + (xv[j] - sv[j]) * mv[j]);
  *(v4h*)(out + i) = o;
}

// ---- fp32 -> f16 convert (weights) -----------------------------------------
__global__ __launch_bounds__(256) void cvt16_kernel(
    const float* __restrict__ src, _Float16* __restrict__ dst)
{
  size_t i = ((size_t)blockIdx.x * blockDim.x + threadIdx.x) * 4;
  v4f s = *(const v4f*)(src + i);
  v4h o;
#pragma unroll
  for (int j = 0; j < 4; ++j) o[j] = (_Float16)s[j];
  *(v4h*)(dst + i) = o;
}

// ---- GEMM: Cout[m,n] = sum_k A16[m,k] * W16[n,k]  (both K-contiguous) ------
// Block: 256 threads = 8 waves, tile 128(M) x 128(N); wave tile 64x32.
// Double-buffered async global->LDS staging, K-stage = 32.
__global__ __launch_bounds__(256) void gemm_nt_wmma_kernel(
    const _Float16* __restrict__ A,   // M x K row-major
    const _Float16* __restrict__ W,   // N x K row-major
    float* __restrict__ Cout,         // M x N row-major
    int M, int N, int K)
{
  constexpr int KSTEP = 32;
  constexpr int LSTR  = 40;          // 32 + 8 halves pad: conflict-free banks
  __shared__ _Float16 lA[2][128 * LSTR];   // 2 x 10 KB
  __shared__ _Float16 lB[2][128 * LSTR];   // 2 x 10 KB

  const int tid  = threadIdx.x;
  const int lane = tid & 31;
  const int wave = tid >> 5;              // 0..7
  const int wm   = wave >> 2;             // 0..1  (M dir)
  const int wn   = wave & 3;              // 0..3  (N dir)
  const int mBlk = blockIdx.x * 128;
  const int nBlk = blockIdx.y * 128;
  const int lh   = lane & 15;             // row within 16x16 tile
  const int hs   = lane >> 4;             // half-select for K chunks

  v8f acc[4][2];
  v8f zero = {};
#pragma unroll
  for (int mi = 0; mi < 4; ++mi)
#pragma unroll
    for (int ni = 0; ni < 2; ++ni) acc[mi][ni] = zero;

  // 128x32 tile = 512 x 16B chunks; thread t copies chunks t and t+256.
  // chunk c: row = c>>2, ko = (c&3)*8 halves.
  auto issue_stage = [&](int bi, int k0) {
#pragma unroll
    for (int q = 0; q < 2; ++q) {
      const int c   = tid + q * 256;
      const int row = c >> 2;
      const int ko  = (c & 3) * 8;
      const unsigned aOff =
          ((unsigned)(mBlk + row) * (unsigned)K + (unsigned)(k0 + ko)) * 2u;
      const unsigned la =
          (unsigned)(unsigned long long)(const void*)&lA[bi][row * LSTR + ko];
      asm volatile("global_load_async_to_lds_b128 %0, %1, %2"
                   :: "v"(la), "v"(aOff), "s"(A) : "memory");
      const unsigned bOff =
          ((unsigned)(nBlk + row) * (unsigned)K + (unsigned)(k0 + ko)) * 2u;
      const unsigned lb =
          (unsigned)(unsigned long long)(const void*)&lB[bi][row * LSTR + ko];
      asm volatile("global_load_async_to_lds_b128 %0, %1, %2"
                   :: "v"(lb), "v"(bOff), "s"(W) : "memory");
    }
  };

  const int nstages = K / KSTEP;
  issue_stage(0, 0);

  for (int s = 0; s < nstages; ++s) {
    const int bi = s & 1;
    if (s + 1 < nstages) {
      issue_stage(bi ^ 1, (s + 1) * KSTEP);
      // 4 newer ops outstanding allowed; in-order => stage-s data has landed
      asm volatile("s_wait_asynccnt 0x4" ::: "memory");
    } else {
      asm volatile("s_wait_asynccnt 0x0" ::: "memory");
    }
    __syncthreads();   // all waves' stage-s copies visible in LDS

    const _Float16* aT = &lA[bi][0];
    const _Float16* bT = &lB[bi][0];

    // A-frag: lane holds row (wm*64+mi*16+lh); K chunks [8*hs,+8) and +16.
    v16h af[4];
#pragma unroll
    for (int mi = 0; mi < 4; ++mi) {
      const _Float16* ap = aT + (wm * 64 + mi * 16 + lh) * LSTR + hs * 8;
      union { v16h v; v8h h[2]; } u;
      u.h[0] = *(const v8h*)(ap);
      u.h[1] = *(const v8h*)(ap + 16);
      af[mi] = u.v;
    }
    // B-frag: lane holds W row (wn*32+ni*16+lh); K chunks [8*hs,+8) and +16.
    v16h bf[2];
#pragma unroll
    for (int ni = 0; ni < 2; ++ni) {
      const _Float16* bp = bT + (wn * 32 + ni * 16 + lh) * LSTR + hs * 8;
      union { v16h v; v8h h[2]; } u;
      u.h[0] = *(const v8h*)(bp);
      u.h[1] = *(const v8h*)(bp + 16);
      bf[ni] = u.v;
    }
#pragma unroll
    for (int mi = 0; mi < 4; ++mi)
#pragma unroll
      for (int ni = 0; ni < 2; ++ni)
        acc[mi][ni] = __builtin_amdgcn_wmma_f32_16x16x32_f16(
            false, af[mi], false, bf[ni], (short)0, acc[mi][ni], false, false);

    __syncthreads();   // reads done before buffer bi is overwritten
  }

  // C/D layout: VGPR r -> row (hs*8 + r), lane lh -> col.
#pragma unroll
  for (int mi = 0; mi < 4; ++mi) {
#pragma unroll
    for (int ni = 0; ni < 2; ++ni) {
      const int col = nBlk + wn * 32 + ni * 16 + lh;
#pragma unroll
      for (int r = 0; r < 8; ++r) {
        const int row = mBlk + wm * 64 + mi * 16 + hs * 8 + r;
        Cout[(size_t)row * N + col] = acc[mi][ni][r];
      }
    }
  }
}

// ---- stabilized WKV elementwise step ---------------------------------------
__global__ __launch_bounds__(256) void wkv_kernel(
    const float* __restrict__ k,  const float* __restrict__ v,
    const float* __restrict__ rp, const float* __restrict__ sA,
    const float* __restrict__ sB, const float* __restrict__ sp,
    const float* __restrict__ x,
    const float* __restrict__ td, const float* __restrict__ tf,
    float* __restrict__ outA, float* __restrict__ outB,
    float* __restrict__ outP, float* __restrict__ outX,
    _Float16* __restrict__ rv16)
{
  size_t i = ((size_t)blockIdx.x * blockDim.x + threadIdx.x) * 4;
  int c = (int)(i & (size_t)(C_DIM - 1));
  v4f kv = *(const v4f*)(k + i);
  v4f vv = *(const v4f*)(v + i);
  v4f rr = *(const v4f*)(rp + i);
  v4f Av = *(const v4f*)(sA + i);
  v4f Bv = *(const v4f*)(sB + i);
  v4f Pv = *(const v4f*)(sp + i);
  v4f Xv = *(const v4f*)(x + i);
  v4f tdv = *(const v4f*)(td + c);
  v4f tfv = *(const v4f*)(tf + c);

  v4f oA, oB, oP; v4h orv;
#pragma unroll
  for (int j = 0; j < 4; ++j) {
    float ww = tfv[j] + kv[j];
    float p  = fmaxf(Pv[j], ww);
    float e1 = __expf(Pv[j] - p);
    float e2 = __expf(ww - p);
    float a  = e1 * Av[j] + e2 * vv[j];
    float b  = e1 * Bv[j] + e2;
    float ww2 = Pv[j] - __expf(tdv[j]);
    float p2  = fmaxf(ww2, kv[j]);
    float e1b = __expf(ww2 - p2);
    float e2b = __expf(kv[j] - p2);
    oA[j] = e1b * Av[j] + e2b * vv[j];
    oB[j] = e1b * Bv[j] + e2b;
    oP[j] = p2;
    float r = 1.0f / (1.0f + __expf(-rr[j]));
    orv[j] = (_Float16)(r * a / b);
  }
  *(v4f*)(outA + i) = oA;
  *(v4f*)(outB + i) = oB;
  *(v4f*)(outP + i) = oP;
  *(v4f*)(outX + i) = Xv;   // new_x = x
  *(v4h*)(rv16 + i) = orv;
}

// ---------------------------------------------------------------------------
extern "C" void kernel_launch(void* const* d_in, const int* in_sizes, int n_in,
                              void* d_out, int out_size, void* d_ws, size_t ws_size,
                              hipStream_t stream) {
  const float* x   = (const float*)d_in[0];
  const float* sA  = (const float*)d_in[1];
  const float* sB  = (const float*)d_in[2];
  const float* sp  = (const float*)d_in[3];
  const float* sx  = (const float*)d_in[4];
  const float* td  = (const float*)d_in[5];
  const float* tf  = (const float*)d_in[6];
  const float* tmk = (const float*)d_in[7];
  const float* tmv = (const float*)d_in[8];
  const float* tmr = (const float*)d_in[9];
  const float* Wk  = (const float*)d_in[10];
  const float* Wv  = (const float*)d_in[11];
  const float* Wr  = (const float*)d_in[12];
  const float* Wo  = (const float*)d_in[13];

  float* out_rwkv = (float*)d_out;
  float* out_A    = out_rwkv + BC;
  float* out_B    = out_A + BC;
  float* out_P    = out_B + BC;
  float* out_X    = out_P + BC;

  // workspace layout (reused sequentially):
  //   mixbuf : BC f16   (also reused as rv buffer for the output GEMM)
  //   w16buf : CxC f16
  //   kbuf, vbuf, rbuf : BC f32 each
  _Float16* mixbuf = (_Float16*)d_ws;
  _Float16* w16buf = mixbuf + BC;
  float* kbuf = (float*)(w16buf + CxC);
  float* vbuf = kbuf + BC;
  float* rbuf = vbuf + BC;

  const int EW_BLOCKS  = (int)(BC / (4 * 256));   // elementwise over B*C
  const int WCV_BLOCKS = (int)(CxC / (4 * 256));  // weight convert
  dim3 gemmGrid(B_DIM / 128, C_DIM / 128);        // 64 x 16

  // --- k = mix_k(x) @ Wk^T ---
  mix_one_kernel<<<EW_BLOCKS, 256, 0, stream>>>(x, sx, tmk, mixbuf);
  cvt16_kernel<<<WCV_BLOCKS, 256, 0, stream>>>(Wk, w16buf);
  gemm_nt_wmma_kernel<<<gemmGrid, 256, 0, stream>>>(mixbuf, w16buf, kbuf,
                                                    B_DIM, C_DIM, C_DIM);
  // --- v = mix_v(x) @ Wv^T ---
  mix_one_kernel<<<EW_BLOCKS, 256, 0, stream>>>(x, sx, tmv, mixbuf);
  cvt16_kernel<<<WCV_BLOCKS, 256, 0, stream>>>(Wv, w16buf);
  gemm_nt_wmma_kernel<<<gemmGrid, 256, 0, stream>>>(mixbuf, w16buf, vbuf,
                                                    B_DIM, C_DIM, C_DIM);
  // --- r_pre = mix_r(x) @ Wr^T ---
  mix_one_kernel<<<EW_BLOCKS, 256, 0, stream>>>(x, sx, tmr, mixbuf);
  cvt16_kernel<<<WCV_BLOCKS, 256, 0, stream>>>(Wr, w16buf);
  gemm_nt_wmma_kernel<<<gemmGrid, 256, 0, stream>>>(mixbuf, w16buf, rbuf,
                                                    B_DIM, C_DIM, C_DIM);
  // --- WKV state math; rv = sigmoid(r_pre)*a/b (f16 into mixbuf) ---
  wkv_kernel<<<EW_BLOCKS, 256, 0, stream>>>(kbuf, vbuf, rbuf, sA, sB, sp, x,
                                            td, tf, out_A, out_B, out_P, out_X,
                                            mixbuf);
  // --- rwkv = rv @ Wo^T ---
  cvt16_kernel<<<WCV_BLOCKS, 256, 0, stream>>>(Wo, w16buf);
  gemm_nt_wmma_kernel<<<gemmGrid, 256, 0, stream>>>(mixbuf, w16buf, out_rwkv,
                                                    B_DIM, C_DIM, C_DIM);
}
